// StatefulBDHProcessor_11811160064724
// MI455X (gfx1250) — matrix operand order repaired and checked by reading, not compiled, not run on hardware
//
#include <hip/hip_runtime.h>
#include <hip/hip_bf16.h>
#include <math.h>

// Problem constants (match reference)
#define D_EMB  512
#define NHEAD  16
#define NLAT   128
#define FLATN  2048      // NHEAD * NLAT
#define TT     512
#define BB     2
#define NCHUNK 8
#define NLAYERS 2
#define LN_EPS 1e-5f
#define THETA_F 10000.0f
#define TWO_PI_F (2.0f * 3.14159f)   // reference uses 2*3.14159 literally

typedef __attribute__((ext_vector_type(16))) _Float16 v16h;
typedef __attribute__((ext_vector_type(8)))  float    v8f;

__device__ __forceinline__ v8f zero8() {
  v8f z = {0.f, 0.f, 0.f, 0.f, 0.f, 0.f, 0.f, 0.f};
  return z;
}

__device__ __forceinline__ v8f wmma_f16(v16h a, v16h b, v8f c) {
  // v_wmma_f32_16x16x32_f16 : D = A(16x32 f16) * B(32x16 f16) + C(f32)
  return __builtin_amdgcn_wmma_f32_16x16x32_f16(false, a, false, b, (short)0, c, false, false);
}

// A fragment (16x32 f16, row-major A[m][k], lda in elements).
// Lanes 0-15: row m, K k0+0..7 (halves 0..7) and k0+16..23 (halves 8..15).
// Lanes 16-31: row m, K k0+8..15 and k0+24..31.
__device__ __forceinline__ v16h load_a_frag(const _Float16* __restrict__ A, int lda,
                                            int row0, int k0, int lane) {
  const _Float16* p = A + (size_t)(row0 + (lane & 15)) * lda + (size_t)(k0 + ((lane >> 4) << 3));
  union { v16h v; uint4 q[2]; } u;
  u.q[0] = *reinterpret_cast<const uint4*>(p);
  u.q[1] = *reinterpret_cast<const uint4*>(p + 16);
  return u.v;
}

// B fragment (32x16 f16) from pre-transposed Bt[n][k] (row-major, ldbt in elements).
// Lanes 0-15: col n, K k0+0..7 and k0+16..23; lanes 16-31: col n, K k0+8..15 and k0+24..31.
__device__ __forceinline__ v16h load_b_frag(const _Float16* __restrict__ Bt, int ldbt,
                                            int col0, int k0, int lane) {
  const _Float16* p = Bt + (size_t)(col0 + (lane & 15)) * ldbt + (size_t)(k0 + ((lane >> 4) << 3));
  union { v16h v; uint4 q[2]; } u;
  u.q[0] = *reinterpret_cast<const uint4*>(p);
  u.q[1] = *reinterpret_cast<const uint4*>(p + 16);
  return u.v;
}

// Wave computes a 32(m) x 64(n) tile: accumulators c[mi*4+ni] cover
// (m0+16*mi, n0+16*ni). 8 WMMA per K-step of 32, fed by 6 fragment loads.
__device__ __forceinline__ void gemm_core(const _Float16* __restrict__ A, const _Float16* __restrict__ Bt,
                                          int lda, int ldbt, int K, int m0, int n0, int lane, v8f c[8]) {
  // Speculative warm-up of the first tiles (global_prefetch_b8 on gfx1250).
  __builtin_prefetch(A + (size_t)(m0 + (lane & 15)) * lda, 0, 3);
  __builtin_prefetch(Bt + (size_t)(n0 + (lane & 15)) * ldbt, 0, 3);
  for (int k = 0; k < K; k += 32) {
    v16h a0 = load_a_frag(A, lda, m0,      k, lane);
    v16h a1 = load_a_frag(A, lda, m0 + 16, k, lane);
    v16h b0 = load_b_frag(Bt, ldbt, n0,      k, lane);
    v16h b1 = load_b_frag(Bt, ldbt, n0 + 16, k, lane);
    v16h b2 = load_b_frag(Bt, ldbt, n0 + 32, k, lane);
    v16h b3 = load_b_frag(Bt, ldbt, n0 + 48, k, lane);
    c[0] = wmma_f16(a0, b0, c[0]);
    c[1] = wmma_f16(a0, b1, c[1]);
    c[2] = wmma_f16(a0, b2, c[2]);
    c[3] = wmma_f16(a0, b3, c[3]);
    c[4] = wmma_f16(a1, b0, c[4]);
    c[5] = wmma_f16(a1, b1, c[5]);
    c[6] = wmma_f16(a1, b2, c[6]);
    c[7] = wmma_f16(a1, b3, c[7]);
  }
}

// Block = 128 threads = 4 waves arranged 2(m) x 2(n); block tile 64(m) x 128(n).
#define TILE_SETUP() \
  const int lane = threadIdx.x & 31; \
  const int wave = threadIdx.x >> 5; \
  const int m0 = blockIdx.y * 64  + (wave >> 1) * 32; \
  const int n0 = blockIdx.x * 128 + (wave & 1) * 64;

// Epilogue element coordinates for accumulator tile ti (0..7):
//   mi = ti>>2, ni = ti&3
//   col  = n0 + 16*ni + (lane&15)
//   row  = m0 + 16*mi + 8*(lane>>4) + vi      (vi = 0..7)

// ---------------------------------------------------------------------------
// GEMM kernels (epilogue variants)
// ---------------------------------------------------------------------------

// x_sparse = relu(x @ enc); xr = rotary(x_sparse); outputs xs(f16), xr(f16),
// xrT(f16, [b][flatn][t]) and atomic ks[b][flatn] += sum_t xr.
__global__ __launch_bounds__(128) void k_gemm_enc(
    const _Float16* __restrict__ A, const _Float16* __restrict__ Bt,
    const float* __restrict__ cosT, const float* __restrict__ sinT,
    _Float16* __restrict__ xs_out, _Float16* __restrict__ xr_out,
    _Float16* __restrict__ xrT_out, float* __restrict__ ks) {
  TILE_SETUP();
  v8f c[8];
#pragma unroll
  for (int i = 0; i < 8; ++i) c[i] = zero8();
  gemm_core(A, Bt, D_EMB, D_EMB, D_EMB, m0, n0, lane, c);
  const int b = m0 >> 9;  // rows of a wave tile never straddle the T=512 boundary
#pragma unroll
  for (int ti = 0; ti < 8; ++ti) {
    const int col  = n0 + ((ti & 3) << 4) + (lane & 15);
    const int nn   = col & (NLAT - 1);
    const int rowb = m0 + ((ti >> 2) << 4) + ((lane >> 4) << 3);
    float ssum = 0.f;
#pragma unroll
    for (int vi = 0; vi < 8; ++vi) {
      const int row = rowb + vi;
      const int t = row & (TT - 1);
      float xs = fmaxf(c[ti][vi], 0.f);
      // rotary pair partner: column j^1 lives in lane^1 (same row, same accumulator)
      float partner = __shfl_xor(xs, 1, 32);
      float rh = ((col & 1) == 0) ? -partner : partner;   // rot_half
      float cs = cosT[t * NLAT + nn];
      float sn = sinT[t * NLAT + nn];
      float xr = xs * cs + rh * sn;
      xs_out[(size_t)row * FLATN + col] = (_Float16)xs;
      xr_out[(size_t)row * FLATN + col] = (_Float16)xr;
      xrT_out[((size_t)b * FLATN + col) * TT + t] = (_Float16)xr;
      ssum += xr;
    }
    atomicAdd(&ks[b * FLATN + col], ssum);
  }
}

// kv[b] (2048x512, f32) += xrT[b] (2048xT) @ x[b] (TxD)   (Bt = xhT[b] = x^T)
// RMW epilogue: WMMA pipeline starts immediately; C tile read overlaps the tail.
__global__ __launch_bounds__(128) void k_gemm_kv(
    const _Float16* __restrict__ xrT, const _Float16* __restrict__ xhT,
    float* __restrict__ kv) {
  TILE_SETUP();
  const int b = blockIdx.z;
  const _Float16* A  = xrT + (size_t)b * FLATN * TT;
  const _Float16* Bt = xhT + (size_t)b * D_EMB * TT;
  float* C = kv + (size_t)b * FLATN * D_EMB;
  v8f c[8];
#pragma unroll
  for (int i = 0; i < 8; ++i) c[i] = zero8();
  gemm_core(A, Bt, TT, TT, TT, m0, n0, lane, c);
#pragma unroll
  for (int ti = 0; ti < 8; ++ti) {
    const int col  = n0 + ((ti & 3) << 4) + (lane & 15);
    const int rowb = m0 + ((ti >> 2) << 4) + ((lane >> 4) << 3);
#pragma unroll
    for (int vi = 0; vi < 8; ++vi) {
      size_t idx = (size_t)(rowb + vi) * D_EMB + col;
      C[idx] = C[idx] + c[ti][vi];
    }
  }
}

// Plain f32-out GEMM with per-z batch strides (used for local-sum and yMLP).
__global__ __launch_bounds__(128) void k_gemm_f32(
    const _Float16* __restrict__ A, const _Float16* __restrict__ Bt,
    float* __restrict__ C, int lda, int ldbt, int ldc, int K,
    long sA, long sB, long sC) {
  TILE_SETUP();
  const int b = blockIdx.z;
  A  += (size_t)b * sA;
  Bt += (size_t)b * sB;
  C  += (size_t)b * sC;
  v8f c[8];
#pragma unroll
  for (int i = 0; i < 8; ++i) c[i] = zero8();
  gemm_core(A, Bt, lda, ldbt, K, m0, n0, lane, c);
#pragma unroll
  for (int ti = 0; ti < 8; ++ti) {
    const int col  = n0 + ((ti & 3) << 4) + (lane & 15);
    const int rowb = m0 + ((ti >> 2) << 4) + ((lane >> 4) << 3);
#pragma unroll
    for (int vi = 0; vi < 8; ++vi) C[(size_t)(rowb + vi) * ldc + col] = c[ti][vi];
  }
}

// y_sparse = relu(yKV @ encv); xy = x_sparse * y_sparse (f16 out);
// layer 0: accumulate sum(xy) per batch into chunkacc (for state_neurons).
__global__ __launch_bounds__(128) void k_gemm_yspr(
    const _Float16* __restrict__ A, const _Float16* __restrict__ Bt,
    const _Float16* __restrict__ xs_in, _Float16* __restrict__ xy_out,
    float* __restrict__ chunkacc, int accTrace) {
  TILE_SETUP();
  v8f c[8];
#pragma unroll
  for (int i = 0; i < 8; ++i) c[i] = zero8();
  gemm_core(A, Bt, D_EMB, D_EMB, D_EMB, m0, n0, lane, c);
  const int b = m0 >> 9;
  float tsum = 0.f;
#pragma unroll
  for (int ti = 0; ti < 8; ++ti) {
    const int col  = n0 + ((ti & 3) << 4) + (lane & 15);
    const int rowb = m0 + ((ti >> 2) << 4) + ((lane >> 4) << 3);
#pragma unroll
    for (int vi = 0; vi < 8; ++vi) {
      const int row = rowb + vi;
      float y = fmaxf(c[ti][vi], 0.f);
      float xy = (float)xs_in[(size_t)row * FLATN + col] * y;
      xy_out[(size_t)row * FLATN + col] = (_Float16)xy;
      tsum += xy;
    }
  }
  if (accTrace) atomicAdd(&chunkacc[b], tsum);
}

// q = e @ wq^T + bq (f16 out). Bt[n][k] = wq[n][k] (wq row-major directly).
__global__ __launch_bounds__(128) void k_gemm_q(
    const _Float16* __restrict__ A, const _Float16* __restrict__ Bt,
    const float* __restrict__ bias, _Float16* __restrict__ O) {
  TILE_SETUP();
  v8f c[8];
#pragma unroll
  for (int i = 0; i < 8; ++i) c[i] = zero8();
  gemm_core(A, Bt, D_EMB, D_EMB, D_EMB, m0, n0, lane, c);
#pragma unroll
  for (int ti = 0; ti < 8; ++ti) {
    const int col  = n0 + ((ti & 3) << 4) + (lane & 15);
    const int rowb = m0 + ((ti >> 2) << 4) + ((lane >> 4) << 3);
    const float bcol = bias[col];
#pragma unroll
    for (int vi = 0; vi < 8; ++vi)
      O[(size_t)(rowb + vi) * D_EMB + col] = (_Float16)(c[ti][vi] + bcol);
  }
}

// q_sparse = relu(q @ enc); accumulate column sums over t into qsum[b][flatn].
__global__ __launch_bounds__(128) void k_gemm_qspr(
    const _Float16* __restrict__ A, const _Float16* __restrict__ Bt,
    float* __restrict__ qsum) {
  TILE_SETUP();
  v8f c[8];
#pragma unroll
  for (int i = 0; i < 8; ++i) c[i] = zero8();
  gemm_core(A, Bt, D_EMB, D_EMB, D_EMB, m0, n0, lane, c);
  const int b = m0 >> 9;
#pragma unroll
  for (int ti = 0; ti < 8; ++ti) {
    const int col = n0 + ((ti & 3) << 4) + (lane & 15);
    float s = 0.f;
#pragma unroll
    for (int vi = 0; vi < 8; ++vi) s += fmaxf(c[ti][vi], 0.f);
    atomicAdd(&qsum[b * FLATN + col], s);
  }
}

// ---------------------------------------------------------------------------
// Elementwise / reduction kernels
// ---------------------------------------------------------------------------

__global__ void k_zero(float* __restrict__ p, size_t n) {
  for (size_t i = (size_t)blockIdx.x * blockDim.x + threadIdx.x; i < n;
       i += (size_t)gridDim.x * blockDim.x)
    p[i] = 0.f;
}

// Convert weights to f16 in B^T layout: enc/encv -> [flatn][d], dec -> [d][flatn], wq -> as-is.
__global__ void k_cvt_weights(const float* __restrict__ enc, const float* __restrict__ encv,
                              const float* __restrict__ dec, const float* __restrict__ wq,
                              _Float16* __restrict__ ench, _Float16* __restrict__ encvh,
                              _Float16* __restrict__ dech, _Float16* __restrict__ wqh) {
  for (size_t idx = (size_t)blockIdx.x * blockDim.x + threadIdx.x;
       idx < (size_t)FLATN * D_EMB; idx += (size_t)gridDim.x * blockDim.x) {
    int j = (int)(idx / D_EMB), d = (int)(idx % D_EMB);
    int h = j >> 7, n = j & (NLAT - 1);
    size_t src = ((size_t)h * D_EMB + d) * NLAT + n;   // encoder[h][d][n]
    ench[idx]  = (_Float16)enc[src];
    encvh[idx] = (_Float16)encv[src];
    int j2 = (int)(idx % FLATN), d2 = (int)(idx / FLATN);
    dech[idx] = (_Float16)dec[(size_t)j2 * D_EMB + d2]; // dech[d][j] = dec[j][d]
    if (idx < (size_t)D_EMB * D_EMB) wqh[idx] = (_Float16)wq[idx];
  }
}

// Rotary cos/sin tables for a chunk: phases[t][n] = ((off+t)*freq[n]) % 1 * 2pi
__global__ void k_rope(float* __restrict__ cosT, float* __restrict__ sinT, float offset) {
  int idx = blockIdx.x * blockDim.x + threadIdx.x;
  if (idx >= TT * NLAT) return;
  int t = idx / NLAT, n = idx % NLAT;
  float f = powf(THETA_F, -(float)(n & ~1) / (float)NLAT);
  float ph = fmodf((offset + (float)t) * f, 1.0f) * TWO_PI_F;
  cosT[idx] = cosf(ph);
  sinT[idx] = sinf(ph);
}

// Embedding gather. full=1 additionally writes f32 x and xhT (x^T f16) for the chunk path.
__global__ void k_gather(const int* __restrict__ ids, const float* __restrict__ embed,
                         float* __restrict__ x, _Float16* __restrict__ xh,
                         _Float16* __restrict__ xhT, int full) {
  size_t idx = (size_t)blockIdx.x * blockDim.x + threadIdx.x;
  if (idx >= (size_t)BB * TT * D_EMB) return;
  int d = (int)(idx % D_EMB);
  int row = (int)(idx / D_EMB);
  int b = row / TT, t = row % TT;
  float v = embed[(size_t)ids[row] * D_EMB + d];
  xh[idx] = (_Float16)v;
  if (full) {
    x[idx] = v;
    xhT[((size_t)b * D_EMB + d) * TT + t] = (_Float16)v;
  }
}

// attn^T f16: attnT[b][d][k] = kv[b][k][d] / max(ks[b][k], 1e-6)
__global__ void k_attn(const float* __restrict__ kv, const float* __restrict__ ks,
                       _Float16* __restrict__ attnT) {
  size_t idx = (size_t)blockIdx.x * blockDim.x + threadIdx.x;
  if (idx >= (size_t)BB * FLATN * D_EMB) return;
  int d = (int)(idx % D_EMB);
  size_t r = idx / D_EMB;
  int k = (int)(r % FLATN);
  int b = (int)(r / FLATN);
  float s = kv[idx] / fmaxf(ks[(size_t)b * FLATN + k], 1e-6f);
  attnT[((size_t)b * D_EMB + d) * FLATN + k] = (_Float16)s;
}

__device__ __forceinline__ void block_reduce2(float& a, float& q, float* s1, float* s2) {
  int tid = threadIdx.x;
  s1[tid] = a; s2[tid] = q;
  __syncthreads();
  for (int off = 128; off > 0; off >>= 1) {
    if (tid < off) { s1[tid] += s1[tid + off]; s2[tid] += s2[tid + off]; }
    __syncthreads();
  }
  a = s1[0]; q = s2[0];
  __syncthreads();
}

// LayerNorm rows of f32 input -> f16 output (yKV path). One block (256 thr) per row, D=512.
__global__ __launch_bounds__(256) void k_ln_rows(const float* __restrict__ in,
                                                 _Float16* __restrict__ outH) {
  __shared__ float s1[256], s2[256];
  int row = blockIdx.x, tid = threadIdx.x;
  const float* r = in + (size_t)row * D_EMB;
  float v0 = r[tid], v1 = r[tid + 256];
  float a = v0 + v1, q = v0 * v0 + v1 * v1;
  block_reduce2(a, q, s1, s2);
  float mean = a / D_EMB;
  float inv = rsqrtf(q / D_EMB - mean * mean + LN_EPS);
  _Float16* o = outH + (size_t)row * D_EMB;
  o[tid]       = (_Float16)((v0 - mean) * inv);
  o[tid + 256] = (_Float16)((v1 - mean) * inv);
}

// x = LN(x + LN(yMLP)); writes x f32, xh f16, xhT f16 (transposed per batch).
__global__ __launch_bounds__(256) void k_lnres(const float* __restrict__ yMLP,
                                               float* __restrict__ x,
                                               _Float16* __restrict__ xh,
                                               _Float16* __restrict__ xhT) {
  __shared__ float s1[256], s2[256];
  int row = blockIdx.x, tid = threadIdx.x;
  int b = row / TT, t = row % TT;
  const float* m = yMLP + (size_t)row * D_EMB;
  float m0 = m[tid], m1 = m[tid + 256];
  float a = m0 + m1, q = m0 * m0 + m1 * m1;
  block_reduce2(a, q, s1, s2);
  float mean = a / D_EMB;
  float inv = rsqrtf(q / D_EMB - mean * mean + LN_EPS);
  float* xr_ = x + (size_t)row * D_EMB;
  float v0 = xr_[tid]       + (m0 - mean) * inv;
  float v1 = xr_[tid + 256] + (m1 - mean) * inv;
  a = v0 + v1; q = v0 * v0 + v1 * v1;
  block_reduce2(a, q, s1, s2);
  mean = a / D_EMB;
  inv = rsqrtf(q / D_EMB - mean * mean + LN_EPS);
  float w0 = (v0 - mean) * inv, w1 = (v1 - mean) * inv;
  xr_[tid] = w0; xr_[tid + 256] = w1;
  xh[(size_t)row * D_EMB + tid]       = (_Float16)w0;
  xh[(size_t)row * D_EMB + tid + 256] = (_Float16)w1;
  xhT[((size_t)b * D_EMB + tid) * TT + t]       = (_Float16)w0;
  xhT[((size_t)b * D_EMB + tid + 256) * TT + t] = (_Float16)w1;
}

// Outputs: embedding [B,D], neuron_trace [B,FLATN], state_neurons [B,FLATN]
__global__ void k_final(const float* __restrict__ kv, const float* __restrict__ ks,
                        const float* __restrict__ qsum, const float* __restrict__ chunkacc,
                        float* __restrict__ out) {
  int idx = blockIdx.x * blockDim.x + threadIdx.x;
  const int n1 = BB * D_EMB;
  const int n2 = n1 + BB * FLATN;
  const int n3 = n2 + BB * FLATN;
  if (idx < n1) {
    int b = idx / D_EMB, d = idx % D_EMB;
    const float* kvb = kv + (size_t)b * FLATN * D_EMB;
    const float* ksb = ks + (size_t)b * FLATN;
    const float* qb  = qsum + (size_t)b * FLATN;
    float acc = 0.f;
    for (int k = 0; k < FLATN; ++k)
      acc += qb[k] * kvb[(size_t)k * D_EMB + d] / fmaxf(ksb[k], 1e-6f);
    out[idx] = acc / (float)(NHEAD * TT);
  } else if (idx < n2) {
    out[idx] = qsum[idx - n1] / (float)TT;
  } else if (idx < n3) {
    int b = (idx - n2) / FLATN;
    // acc recurrence with alpha=1/(c+1) == plain mean of per-chunk means
    out[idx] = chunkacc[b] / ((float)NCHUNK * (float)TT * (float)FLATN);
  }
}

// ---------------------------------------------------------------------------
// Host launch
// ---------------------------------------------------------------------------
extern "C" void kernel_launch(void* const* d_in, const int* in_sizes, int n_in,
                              void* d_out, int out_size, void* d_ws, size_t ws_size,
                              hipStream_t stream) {
  const float* embed = (const float*)d_in[0];
  const float* enc   = (const float*)d_in[1];
  const float* encv  = (const float*)d_in[2];
  const float* dec   = (const float*)d_in[3];
  const float* wq    = (const float*)d_in[4];
  const float* bq    = (const float*)d_in[5];
  const int*   chunks = (const int*)d_in[6];   // [NCHUNK][B][T]
  const int*   backs  = (const int*)d_in[7];   // [B][T]

  char* ws = (char*)d_ws;
  size_t off = 0;
  auto alloc = [&](size_t bytes) -> char* {
    char* p = ws + off;
    off = (off + bytes + 255) & ~(size_t)255;
    return p;
  };

  // State (zeroed every launch, accumulated across chunks/layers) — placed first.
  float* kv       = (float*)alloc((size_t)BB * FLATN * D_EMB * 4);
  float* ks       = (float*)alloc((size_t)BB * FLATN * 4);
  float* qsum     = (float*)alloc((size_t)BB * FLATN * 4);
  float* chunkacc = (float*)alloc(256);
  size_t zeroFloats = off / 4;

  _Float16* ench  = (_Float16*)alloc((size_t)FLATN * D_EMB * 2);
  _Float16* encvh = (_Float16*)alloc((size_t)FLATN * D_EMB * 2);
  _Float16* dech  = (_Float16*)alloc((size_t)D_EMB * FLATN * 2);
  _Float16* wqh   = (_Float16*)alloc((size_t)D_EMB * D_EMB * 2);
  float* cosT = (float*)alloc((size_t)TT * NLAT * 4);
  float* sinT = (float*)alloc((size_t)TT * NLAT * 4);
  float*     x    = (float*)alloc((size_t)BB * TT * D_EMB * 4);
  _Float16*  xh   = (_Float16*)alloc((size_t)BB * TT * D_EMB * 2);
  _Float16*  xhT  = (_Float16*)alloc((size_t)BB * D_EMB * TT * 2);
  _Float16*  xs   = (_Float16*)alloc((size_t)BB * TT * FLATN * 2);
  _Float16*  xrb  = (_Float16*)alloc((size_t)BB * TT * FLATN * 2);
  _Float16*  xrT  = (_Float16*)alloc((size_t)BB * FLATN * TT * 2);
  _Float16*  attnT = (_Float16*)alloc((size_t)BB * D_EMB * FLATN * 2);
  float*     lsum = (float*)alloc((size_t)BB * TT * D_EMB * 4);
  _Float16*  yKVh = (_Float16*)alloc((size_t)BB * TT * D_EMB * 2);
  _Float16*  xy   = (_Float16*)alloc((size_t)BB * TT * FLATN * 2);
  float*     yMLP = (float*)alloc((size_t)BB * TT * D_EMB * 4);
  _Float16*  qh   = (_Float16*)alloc((size_t)BB * TT * D_EMB * 2);
  (void)ws_size; (void)in_sizes; (void)n_in; (void)out_size;

  const int ROWS = BB * TT;  // 1024 (both batches concatenated along M)

  k_zero<<<2048, 256, 0, stream>>>((float*)d_ws, zeroFloats);
  k_cvt_weights<<<4096, 256, 0, stream>>>(enc, encv, dec, wq, ench, encvh, dech, wqh);

  for (int c = 0; c < NCHUNK; ++c) {
    k_rope<<<(TT * NLAT) / 256, 256, 0, stream>>>(cosT, sinT, (float)(c * TT));
    k_gather<<<(ROWS * D_EMB) / 256, 256, 0, stream>>>(chunks + (size_t)c * ROWS, embed,
                                                       x, xh, xhT, 1);
    for (int l = 0; l < NLAYERS; ++l) {
      // x_sparse / xr / xrT / ks
      k_gemm_enc<<<dim3(FLATN / 128, ROWS / 64, 1), 128, 0, stream>>>(
          xh, ench, cosT, sinT, xs, xrb, xrT, ks);
      // kv += xr^T @ x
      k_gemm_kv<<<dim3(D_EMB / 128, FLATN / 64, BB), 128, 0, stream>>>(xrT, xhT, kv);
      // attn = kv / max(ks,1e-6) -> f16 transposed
      k_attn<<<(BB * FLATN * D_EMB) / 256, 256, 0, stream>>>(kv, ks, attnT);
      // sum_h local = xr @ attn  (flat over heads)
      k_gemm_f32<<<dim3(D_EMB / 128, TT / 64, BB), 128, 0, stream>>>(
          xrb, attnT, lsum, FLATN, FLATN, D_EMB, FLATN,
          (long)TT * FLATN, (long)D_EMB * FLATN, (long)TT * D_EMB);
      // yKV = LN(mean_h local)  (LN is scale-invariant, skip /NH)
      k_ln_rows<<<ROWS, 256, 0, stream>>>(lsum, yKVh);
      // y_sparse, xy, (layer0) trace scalar
      k_gemm_yspr<<<dim3(FLATN / 128, ROWS / 64, 1), 128, 0, stream>>>(
          yKVh, encvh, xs, xy, chunkacc, (l == 0) ? 1 : 0);
      // yMLP = xy @ decoder
      k_gemm_f32<<<dim3(D_EMB / 128, ROWS / 64, 1), 128, 0, stream>>>(
          xy, dech, yMLP, FLATN, FLATN, D_EMB, FLATN, 0, 0, 0);
      // x = LN(x + LN(yMLP))
      k_lnres<<<ROWS, 256, 0, stream>>>(yMLP, x, xh, xhT);
    }
  }

  // Query phase
  k_gather<<<(ROWS * D_EMB) / 256, 256, 0, stream>>>(backs, embed, x, xh, xhT, 0);
  k_gemm_q<<<dim3(D_EMB / 128, ROWS / 64, 1), 128, 0, stream>>>(xh, wqh, bq, qh);
  k_gemm_qspr<<<dim3(FLATN / 128, ROWS / 64, 1), 128, 0, stream>>>(qh, ench, qsum);
  k_final<<<(BB * D_EMB + 2 * BB * FLATN + 255) / 256, 256, 0, stream>>>(
      kv, ks, qsum, chunkacc, (float*)d_out);
}
